// G_ATT_2388001816920
// MI455X (gfx1250) — compile-verified, matrix operand order checked by reading, not run on hardware
//
#include <hip/hip_runtime.h>

// ---------------------------------------------------------------------------
// Problem constants (from reference)
// ---------------------------------------------------------------------------
constexpr int B_DIM  = 256;
constexpr int L_DIM  = 512;
constexpr int E_DIM  = 256;
constexpr int TWO_E  = 512;
constexpr int N_ITER = 50;
constexpr int M_ROWS = B_DIM * L_DIM;     // 131072 flattened GEMM rows

// WMMA tiling
constexpr int TILE_ROWS  = 128;           // C rows staged per workgroup (8 waves x 16)
constexpr int LDS_STRIDE = 520;           // bf16 elems per row: 1040 B = 65*16 (aligned), 260 dwords (260%64=4)
constexpr int BTILE_ROWS = 64;            // Wg1 rows (o-channels) staged per o-group
constexpr int C_TILE_ELEMS = TILE_ROWS * LDS_STRIDE;     // ushort elems
constexpr int B_TILE_ELEMS = BTILE_ROWS * LDS_STRIDE;    // ushort elems

typedef __attribute__((ext_vector_type(16))) __bf16 v16bf;
typedef __attribute__((ext_vector_type(8)))  float  v8f;

union Frag {
    v16bf bf;
    unsigned int u[8];
    uint4 q[2];
};

// round-to-nearest-even f32 -> bf16 (bit pattern)
__device__ __forceinline__ unsigned short f2bf(float x) {
    unsigned int u = __float_as_uint(x);
    u += 0x7FFFu + ((u >> 16) & 1u);
    return (unsigned short)(u >> 16);
}

// ---------------------------------------------------------------------------
// Prep kernel A: convert Wg1 (f32 [512,512]) -> bf16 copy in workspace
// ---------------------------------------------------------------------------
__global__ __launch_bounds__(256)
void k_w1_to_bf16(const float* __restrict__ w1, unsigned short* __restrict__ w1bf) {
    int i = blockIdx.x * 256 + threadIdx.x;       // 1024 blocks cover 262144
    w1bf[i] = f2bf(w1[i]);
}

// ---------------------------------------------------------------------------
// Prep kernel B: qterm[b][o] = bias[o] + sum_e C_target[b,e] * Wg2[o,e]
// ---------------------------------------------------------------------------
__global__ __launch_bounds__(256)
void k_qterm(const float* __restrict__ c_target, const float* __restrict__ wg2,
             const float* __restrict__ bias, float* __restrict__ qterm) {
    __shared__ float ct[TWO_E];
    const int b = blockIdx.x;
    const int t = threadIdx.x;
    ct[t]       = c_target[b * TWO_E + t];
    ct[t + 256] = c_target[b * TWO_E + t + 256];
    __syncthreads();

    #pragma unroll
    for (int j = 0; j < 2; ++j) {
        const int o = t + j * 256;
        const float4* w = (const float4*)(wg2 + (size_t)o * TWO_E);
        float acc = bias[o];
        #pragma unroll 4
        for (int e4 = 0; e4 < TWO_E / 4; ++e4) {
            float4 wv = w[e4];
            acc += ct[e4 * 4 + 0] * wv.x + ct[e4 * 4 + 1] * wv.y +
                   ct[e4 * 4 + 2] * wv.z + ct[e4 * 4 + 3] * wv.w;
        }
        qterm[b * TWO_E + o] = acc;
    }
}

// ---------------------------------------------------------------------------
// Main WMMA kernel: score[m] = sum_o relu( (C . Wg1^T)[m,o] + qterm[b,o] ) * Wg0[o]
// 256 threads = 8 waves; wave w owns rows [row0 + 16w, row0 + 16w + 15].
// Wg1 o-group tile (64 ch) staged in LDS, shared by all 8 waves.
// ---------------------------------------------------------------------------
__global__ __launch_bounds__(256)
void k_score_wmma(const float* __restrict__ C, const unsigned short* __restrict__ w1bf,
                  const float* __restrict__ qterm, const float* __restrict__ wg0,
                  float* __restrict__ score) {
    extern __shared__ unsigned short smem[];                 // C tile [128][520] ++ B tile [64][520]
    unsigned short* csm = smem;
    unsigned short* bsm = smem + C_TILE_ELEMS;

    const int t    = threadIdx.x;
    const int lane = t & 31;
    const int wave = t >> 5;
    const size_t row0 = (size_t)blockIdx.x * TILE_ROWS;
    const int b = (int)(row0 >> 9);                          // TILE_ROWS divides L

    // ---- stage C tile (128 x 512 f32) into LDS as bf16, padded stride ----
    #pragma unroll 4
    for (int it = 0; it < 64; ++it) {
        const int flat = it * 1024 + t * 4;                  // covers 65536 elems
        const int r = flat >> 9;
        const int c = flat & 511;
        const float4 v = *(const float4*)(C + (row0 + r) * TWO_E + c);
        uint2 p;
        p.x = (unsigned int)f2bf(v.x) | ((unsigned int)f2bf(v.y) << 16);
        p.y = (unsigned int)f2bf(v.z) | ((unsigned int)f2bf(v.w) << 16);
        *(uint2*)(&csm[r * LDS_STRIDE + c]) = p;             // 8B aligned (1040r + 2c)
    }

    const int half = lane >> 4;         // K-half selector (ISA 7.12.2, 16-bit frag 16x32)
    const int ln   = lane & 15;         // M (for A) / N (for B) index within tile
    const int mRow = wave * 16 + ln;    // LDS row this lane reads for A

    // per-lane 16B-aligned fragment base pointers; all further offsets constant
    const unsigned short* aBase = csm + mRow * LDS_STRIDE + half * 8;
    const unsigned short* bBase = bsm + ln   * LDS_STRIDE + half * 8;

    float sp[8];                        // score partials: rows m = r + 8*half
    #pragma unroll
    for (int r = 0; r < 8; ++r) sp[r] = 0.0f;

    // 32 o-chunks in 8 groups of 4; Wg1 group tile staged in LDS each group
    for (int og = 0; og < 8; ++og) {
        __syncthreads();                // previous B tile fully consumed (also covers C staging)
        // ---- stage B tile: Wg1 bf16 rows [og*64, og*64+63], 64 KB ----
        {
            const unsigned short* src = w1bf + (size_t)og * BTILE_ROWS * TWO_E;
            #pragma unroll
            for (int it = 0; it < 16; ++it) {
                const int flat = it * 1024 + t * 4;          // dword index, 16384 total
                const int r = flat >> 8;                     // 256 dwords per row
                const int c2 = (flat & 255) * 2;             // bf16 element col
                const uint4 v = *(const uint4*)(src + (size_t)r * TWO_E + c2);
                *(uint4*)(&bsm[r * LDS_STRIDE + c2]) = v;
            }
        }
        __syncthreads();

        // prefetch next o-group's Wg1 tile into cache (global_prefetch_b8)
        if (og < 7) {
            const unsigned short* nxt = w1bf + (size_t)(og + 1) * BTILE_ROWS * TWO_E;
            __builtin_prefetch(nxt + (size_t)t * 128, 0, 1);  // 256 threads x 256B = 64 KB
        }

        v8f acc[4] = {};
        #pragma unroll 2
        for (int kc = 0; kc < 16; ++kc) {
            const int k0 = kc * 32;                          // element offset (const)
            // A fragment: two b128 LDS loads
            Frag A;
            A.q[0] = *(const uint4*)(aBase + k0);
            A.q[1] = *(const uint4*)(aBase + k0 + 16);
            // All 4 B fragments up front: 8 b128 LDS loads with constant offsets
            Frag B0, B1, B2, B3;
            B0.q[0] = *(const uint4*)(bBase + k0);
            B0.q[1] = *(const uint4*)(bBase + k0 + 16);
            B1.q[0] = *(const uint4*)(bBase + 16 * LDS_STRIDE + k0);
            B1.q[1] = *(const uint4*)(bBase + 16 * LDS_STRIDE + k0 + 16);
            B2.q[0] = *(const uint4*)(bBase + 32 * LDS_STRIDE + k0);
            B2.q[1] = *(const uint4*)(bBase + 32 * LDS_STRIDE + k0 + 16);
            B3.q[0] = *(const uint4*)(bBase + 48 * LDS_STRIDE + k0);
            B3.q[1] = *(const uint4*)(bBase + 48 * LDS_STRIDE + k0 + 16);
            acc[0] = __builtin_amdgcn_wmma_f32_16x16x32_bf16(
                false, A.bf, false, B0.bf, (short)0, acc[0], false, false);
            acc[1] = __builtin_amdgcn_wmma_f32_16x16x32_bf16(
                false, A.bf, false, B1.bf, (short)0, acc[1], false, false);
            acc[2] = __builtin_amdgcn_wmma_f32_16x16x32_bf16(
                false, A.bf, false, B2.bf, (short)0, acc[2], false, false);
            acc[3] = __builtin_amdgcn_wmma_f32_16x16x32_bf16(
                false, A.bf, false, B3.bf, (short)0, acc[3], false, false);
        }
        // epilogue: + qterm, relu, * Wg0, accumulate per-row score partials
        #pragma unroll
        for (int oi = 0; oi < 4; ++oi) {
            const int o = (og * 4 + oi) * 16 + ln;
            const float qv = qterm[b * TWO_E + o];
            const float w0 = wg0[o];
            #pragma unroll
            for (int r = 0; r < 8; ++r) {
                float hv = acc[oi][r] + qv;
                hv = fmaxf(hv, 0.0f);
                sp[r] += hv * w0;
            }
        }
    }

    // reduce over the 16 N-lanes (wave32: halves stay independent under xor<16)
    #pragma unroll
    for (int off = 1; off < 16; off <<= 1) {
        #pragma unroll
        for (int r = 0; r < 8; ++r)
            sp[r] += __shfl_xor(sp[r], off, 32);
    }
    if (ln == 0) {
        const size_t base = row0 + wave * 16 + half * 8;     // C/D: VGPR r -> M = r + 8*half
        #pragma unroll
        for (int r = 0; r < 8; ++r)
            score[base + r] = sp[r];
    }
}

// ---------------------------------------------------------------------------
// Entmax bisection (50 iters) over L per batch, then out[b,d] = sum_l p[l]*X[b,l,d]
// ---------------------------------------------------------------------------
__device__ __forceinline__ float pfun(float z, float inv) {
    return powf(fmaxf(z, 0.0f), inv);
}

__global__ __launch_bounds__(256)
void k_entmax_out(const float* __restrict__ score, const int* __restrict__ mask,
                  const float* __restrict__ alpha, const float* __restrict__ X,
                  float* __restrict__ out) {
    __shared__ float xs[L_DIM];
    __shared__ float pn[L_DIM];
    __shared__ float red[256];
    __shared__ float bc;

    const int b = blockIdx.x;
    const int t = threadIdx.x;
    const float am1 = alpha[b] - 1.0f;
    const float inv = 1.0f / am1;

    #pragma unroll
    for (int j = 0; j < 2; ++j) {
        const int l = t + j * 256;
        float s = score[b * L_DIM + l];
        if (mask[b * L_DIM + l] == 0) s = -__builtin_inff();
        xs[l] = s * am1;
    }
    __syncthreads();

    // max over row
    red[t] = fmaxf(xs[t], xs[t + 256]);
    __syncthreads();
    for (int s = 128; s > 0; s >>= 1) {
        if (t < s) red[t] = fmaxf(red[t], red[t + s]);
        __syncthreads();
    }
    if (t == 0) bc = red[0];
    __syncthreads();
    const float mx = bc;
    __syncthreads();

    float tau_lo = mx - 1.0f;
    const float tau_hi = mx - powf(1.0f / (float)L_DIM, am1);

    // f_lo
    red[t] = pfun(xs[t] - tau_lo, inv) + pfun(xs[t + 256] - tau_lo, inv);
    __syncthreads();
    for (int s = 128; s > 0; s >>= 1) {
        if (t < s) red[t] += red[t + s];
        __syncthreads();
    }
    if (t == 0) bc = red[0];
    __syncthreads();
    const float f_lo = bc - 1.0f;
    __syncthreads();

    float dm = tau_hi - tau_lo;
    float tau_m = tau_lo;
    for (int it = 0; it < N_ITER; ++it) {
        dm *= 0.5f;
        tau_m = tau_lo + dm;
        red[t] = pfun(xs[t] - tau_m, inv) + pfun(xs[t + 256] - tau_m, inv);
        __syncthreads();
        for (int s = 128; s > 0; s >>= 1) {
            if (t < s) red[t] += red[t + s];
            __syncthreads();
        }
        if (t == 0) bc = red[0];
        __syncthreads();
        const float f_m = bc - 1.0f;
        __syncthreads();
        if (f_m * f_lo >= 0.0f) tau_lo = tau_m;   // identical on all threads
    }

    // final p, normalization
    const float p0 = pfun(xs[t] - tau_m, inv);
    const float p1 = pfun(xs[t + 256] - tau_m, inv);
    red[t] = p0 + p1;
    __syncthreads();
    for (int s = 128; s > 0; s >>= 1) {
        if (t < s) red[t] += red[t + s];
        __syncthreads();
    }
    if (t == 0) bc = red[0];
    __syncthreads();
    const float invZ = 1.0f / bc;
    pn[t]       = p0 * invZ;
    pn[t + 256] = p1 * invZ;
    __syncthreads();

    // out[b,d] = sum_l pn[l] * X[b,l,d]  (coalesced across d = t)
    const float* Xb = X + (size_t)b * L_DIM * E_DIM;
    float acc = 0.0f;
    #pragma unroll 4
    for (int l = 0; l < L_DIM; ++l)
        acc += pn[l] * Xb[(size_t)l * E_DIM + t];
    out[b * E_DIM + t] = acc;
}

// ---------------------------------------------------------------------------
// Launcher
// ---------------------------------------------------------------------------
extern "C" void kernel_launch(void* const* d_in, const int* in_sizes, int n_in,
                              void* d_out, int out_size, void* d_ws, size_t ws_size,
                              hipStream_t stream) {
    const float* C_target = (const float*)d_in[0];
    const float* C        = (const float*)d_in[1];
    const float* X        = (const float*)d_in[2];
    const float* alpha    = (const float*)d_in[3];
    const int*   mask     = (const int*)  d_in[4];
    const float* Wg1      = (const float*)d_in[5];
    const float* Wg2      = (const float*)d_in[6];
    const float* Wg0      = (const float*)d_in[7];
    const float* bias     = (const float*)d_in[8];
    float* out = (float*)d_out;

    // workspace layout
    char* ws = (char*)d_ws;
    unsigned short* w1bf = (unsigned short*)ws;                  // 512 KB
    float* qterm = (float*)(ws + (size_t)TWO_E * TWO_E * 2);     // 512 KB
    float* score = qterm + (size_t)B_DIM * TWO_E;                // 512 KB

    // prep
    k_w1_to_bf16<<<(TWO_E * TWO_E) / 256, 256, 0, stream>>>(Wg1, w1bf);
    k_qterm<<<B_DIM, 256, 0, stream>>>(C_target, Wg2, bias, qterm);

    // fused GEMM -> relu -> dot(Wg0) via WMMA; ~195 KB dynamic LDS per WG
    const size_t lds_bytes = (size_t)(C_TILE_ELEMS + B_TILE_ELEMS) * sizeof(unsigned short);
    k_score_wmma<<<M_ROWS / TILE_ROWS, 256, lds_bytes, stream>>>(C, w1bf, qterm, Wg0, score);

    // entmax bisection + weighted sum over X
    k_entmax_out<<<B_DIM, 256, 0, stream>>>(score, mask, alpha, X, out);
}
